// GCNTime_33921651704094
// MI455X (gfx1250) — matrix operand order repaired and checked by reading, not compiled, not run on hardware
//
#include <hip/hip_runtime.h>
#include <hip/hip_bf16.h>

// ---------------------------------------------------------------------------
// Problem constants (match reference)
// ---------------------------------------------------------------------------
#define NN    2048
#define TT    16
#define HH    128
#define DFF_  2048
#define EE    32768
#define LL    2

typedef __attribute__((ext_vector_type(16))) _Float16 v16h;
typedef __attribute__((ext_vector_type(8)))  _Float16 v8h;
typedef __attribute__((ext_vector_type(4)))  _Float16 v4h;
typedef __attribute__((ext_vector_type(8)))  float    v8f;
typedef __attribute__((ext_vector_type(4)))  float    v4f;

__device__ __forceinline__ v8f v8f_zero() {
  v8f z;
#pragma unroll
  for (int i = 0; i < 8; ++i) z[i] = 0.f;
  return z;
}

__device__ __forceinline__ v4h cvt4(v4f x) {
  v4h h;
#pragma unroll
  for (int i = 0; i < 4; ++i) h[i] = (_Float16)x[i];
  return h;
}

__device__ __forceinline__ v8f wmma16(v16h a, v16h b, v8f c) {
  // D = A(16x32 f16) * B(32x16 f16) + C(16x16 f32)
  return __builtin_amdgcn_wmma_f32_16x16x32_f16(false, a, false, b, (short)0, c,
                                                false, false);
}

// ---------------------------------------------------------------------------
// Fragment-major LDS swizzles (inverse of CDNA5 WMMA VGPR layouts).
// Each 16x32 (A) / 32x16 (B) subtile occupies 512 halves = 32 lanes x 16
// contiguous halves; a fragment load is then one aligned 32-byte LDS read.
// A-swizzle maps 4-aligned k-runs to 4 contiguous slots (vectorizable stores);
// B-swizzle maps k to slot 8*(k>>4)+(k&7) in lane half (k>>3)&1, n to lane&15.
// ---------------------------------------------------------------------------
__device__ __forceinline__ int a_sw(int m, int k) {      // m in [0,16), k in [0,32)
  int hi   = (k >> 3) & 1;
  int lane = m + 16 * hi;
  int g    = ((k >> 4) << 2) + ((k & 7) >> 1);
  return lane * 16 + 2 * g + (k & 1);
}
__device__ __forceinline__ int b_sw(int k, int n) {      // k in [0,32), n in [0,16)
  int lane = n + 16 * ((k >> 3) & 1);
  return lane * 16 + ((k >> 4) << 3) + (k & 7);
}
__device__ __forceinline__ v16h load_frag(const _Float16* grp) {
  return *(const v16h*)(grp + (threadIdx.x & 31) * 16);
}

// ---------------------------------------------------------------------------
// Weight conversion kernels
// ---------------------------------------------------------------------------
__global__ void k_convert_f16(const float* __restrict__ W, _Float16* __restrict__ O, int n) {
  int i = blockIdx.x * blockDim.x + threadIdx.x;
  if (i < n) O[i] = (_Float16)W[i];
}

// out[c*R + r] = W[r*C + c]  :  [R,C] -> [C,R]
__global__ void k_transpose_f16(const float* __restrict__ W, _Float16* __restrict__ O,
                                int R, int C) {
  int i = blockIdx.x * blockDim.x + threadIdx.x;
  if (i >= R * C) return;
  int r = i / C, c = i - r * C;
  O[(size_t)c * R + r] = (_Float16)W[i];
}

// ---------------------------------------------------------------------------
// Degree / normalization
// ---------------------------------------------------------------------------
__global__ void k_deg_init(float* deg, int n) {
  int i = blockIdx.x * blockDim.x + threadIdx.x;
  if (i < n) deg[i] = 1.0f;  // self loop
}
__global__ void k_deg_edges(const int* __restrict__ dst, float* deg, int e) {
  int i = blockIdx.x * blockDim.x + threadIdx.x;
  if (i < e) atomicAdd(&deg[dst[i]], 1.0f);
}
__global__ void k_deg_rsqrt(float* deg, int n) {
  int i = blockIdx.x * blockDim.x + threadIdx.x;
  if (i < n) deg[i] = rsqrtf(deg[i]);
}

// ---------------------------------------------------------------------------
// GCN aggregation (self loop init + edge scatter)
// ---------------------------------------------------------------------------
__global__ void k_agg_self(const float* __restrict__ hlin, const float* __restrict__ dinv,
                           float* __restrict__ hagg) {
  int i = blockIdx.x * blockDim.x + threadIdx.x;
  if (i >= NN * TT * HH / 4) return;
  int e = i * 4;
  int n = e / (TT * HH);
  float d = dinv[n];
  v4f h = *(const v4f*)(hlin + e);
  *(v4f*)(hagg + e) = h * (d * d);
}

__global__ void k_agg_edges(const float* __restrict__ hlin, const int* __restrict__ src,
                            const int* __restrict__ dst, const float* __restrict__ dinv,
                            float* __restrict__ hagg) {
  int e = blockIdx.x;
  int s = src[e], d = dst[e];
  float w = dinv[s] * dinv[d];
  const float* in = hlin + (size_t)s * (TT * HH);
  float* out = hagg + (size_t)d * (TT * HH);
  for (int i = threadIdx.x * 4; i < TT * HH; i += blockDim.x * 4) {
    v4f x = *(const v4f*)(in + i);
#pragma unroll
    for (int j = 0; j < 4; ++j) atomicAdd(&out[i + j], x[j] * w);
  }
}

// ---------------------------------------------------------------------------
// bias + relu + *sqrt(H) + positional encoding, and transpose [N,T,H]->[T,N,H]
// ---------------------------------------------------------------------------
__device__ __forceinline__ float pe_val(int t, int c) {
  float inv = powf(10000.f, -2.f * (float)c / (float)HH);
  float arg = (float)t * inv;
  return (c & 1) ? cosf(arg) : sinf(arg);
}

__global__ void k_relu_pe_y(const float* __restrict__ hagg, const float* __restrict__ bias,
                            float* __restrict__ y) {
  int i = blockIdx.x * blockDim.x + threadIdx.x;
  if (i >= NN * TT * HH / 4) return;
  int e = i * 4;
  int n = e / (TT * HH);
  int r = e - n * (TT * HH);
  int t = r >> 7;
  int c = r & 127;
  v4f h  = *(const v4f*)(hagg + e);
  v4f bb = *(const v4f*)(bias + c);
  v4f o;
#pragma unroll
  for (int j = 0; j < 4; ++j)
    o[j] = fmaxf(h[j] + bb[j], 0.f) * 11.3137084989847604f + pe_val(t, c + j);
  *(v4f*)(y + ((size_t)t * NN + n) * HH + c) = o;
}

__global__ void k_trans_to_x(const float* __restrict__ y, float* __restrict__ x) {
  int i = blockIdx.x * blockDim.x + threadIdx.x;
  if (i >= NN * TT * HH / 4) return;
  int e = i * 4;
  int t = e / (NN * HH);
  int r = e - t * (NN * HH);
  int n = r >> 7;
  int c = r & 127;
  *(v4f*)(x + ((size_t)n * TT + t) * HH + c) = *(const v4f*)(y + e);
}

// ---------------------------------------------------------------------------
// Generic WMMA GEMM:  C[b] = A[b](MxK,f32) * B(KxN,f16) (+bias) (+resid[b])
// Block tile 64x64, 8 waves, BK=32. All dims multiples of tile sizes.
// ---------------------------------------------------------------------------
template <bool BIAS, bool RESID>
__global__ void k_gemm(const float* __restrict__ A, long long sA, int lda,
                       const _Float16* __restrict__ B,
                       const float* __restrict__ bias,
                       const float* __restrict__ resid, long long sR,
                       float* __restrict__ C, long long sC, int ldc,
                       int M, int Nn, int K) {
  __shared__ alignas(32) _Float16 As[4 * 512];  // 4 m-subtiles of 16x32
  __shared__ alignas(32) _Float16 Bs[4 * 512];  // 4 n-subtiles of 32x16
  A += (long long)blockIdx.z * sA;
  C += (long long)blockIdx.z * sC;
  if (RESID) resid += (long long)blockIdx.z * sR;

  int tid  = threadIdx.x;
  int wave = tid >> 5;
  int lane = tid & 31;
  int ln = lane & 15, lhi = lane >> 4;
  int mt  = wave >> 1;
  int ntb = (wave & 1) * 2;

  v8f acc0 = v8f_zero(), acc1 = v8f_zero();
  int row0 = blockIdx.y * 64;
  int col0 = blockIdx.x * 64;

  for (int kb = 0; kb < K; kb += 32) {
    __syncthreads();
    // A tile 64x32: float4 loads, contiguous v4h LDS stores
    for (int i = tid; i < 512; i += 256) {
      int r = i >> 3, c4 = (i & 7) * 4;
      v4f x = *(const v4f*)(A + (size_t)(row0 + r) * lda + kb + c4);
      *(v4h*)(As + (r >> 4) * 512 + a_sw(r & 15, c4)) = cvt4(x);
    }
    // B tile 32x64: v8h global loads, lane-strided b16 stores
    {
      int i = tid;
      int k = i >> 3, n8 = (i & 7) * 8;
      v8h x = *(const v8h*)(B + (size_t)(kb + k) * Nn + col0 + n8);
#pragma unroll
      for (int j = 0; j < 8; ++j) {
        int c = n8 + j;
        Bs[(c >> 4) * 512 + b_sw(k, c & 15)] = x[j];
      }
    }
    __syncthreads();
    v16h a  = load_frag(As + mt * 512);
    v16h b0 = load_frag(Bs + ntb * 512);
    v16h b1 = load_frag(Bs + (ntb + 1) * 512);
    acc0 = wmma16(a, b0, acc0);
    acc1 = wmma16(a, b1, acc1);
  }

#pragma unroll
  for (int j = 0; j < 2; ++j) {
    v8f& acc = j ? acc1 : acc0;
    int col = col0 + (ntb + j) * 16 + ln;
    float bb = BIAS ? bias[col] : 0.f;
#pragma unroll
    for (int r = 0; r < 8; ++r) {
      size_t row = (size_t)(row0 + mt * 16 + r + 8 * lhi);
      float v = acc[r] + bb;
      if (RESID) v += resid[row * ldc + col];
      C[row * ldc + col] = v;
    }
  }
}

// ---------------------------------------------------------------------------
// Flash-style causal attention over the node axis (seq=NN, batch=TT, 1 head).
// qkv: [T,N,3H] f32 (q|k|v). out: [T,N,H] f32.
// Block: 256 threads, BQ=32 query rows, key blocks of 64, kb <= qb (causal).
// ---------------------------------------------------------------------------
__global__ void k_attn(const float* __restrict__ qkv, float* __restrict__ out) {
  // fragment-major groups of 512 halves each
  __shared__ alignas(32) _Float16 Qs[8 * 512];    // A: (mt 0..1) x (kc 0..3)
  __shared__ alignas(32) _Float16 KsT[16 * 512];  // B=K^T: (nt 0..3) x (kc 0..3)
  __shared__ alignas(32) _Float16 Vs[16 * 512];   // B: (nt 0..7) x (kc2 0..1)
  __shared__ alignas(32) _Float16 Ps[4 * 512];    // A: (mt 0..1) x (kc2 0..1)
  __shared__ float Ss[32 * 64];
  __shared__ float mrow[32], lrow[32], srow[32], part[32 * 8];

  int t  = blockIdx.y;
  int qb = blockIdx.x;
  const float* base = qkv + (size_t)t * NN * (3 * HH);
  int tid = threadIdx.x;
  int wave = tid >> 5, lane = tid & 31;
  int ln = lane & 15, lhi = lane >> 4;
  const float scale = 0.0883883476483184f; // 1/sqrt(128)

  // load Q block (32 rows) into A-swizzled groups (float4 -> v4h)
  for (int i = tid; i < 1024; i += 256) {
    int r = i >> 5, f4 = (i & 31) * 4;
    v4f x = *(const v4f*)(base + (size_t)(qb * 32 + r) * (3 * HH) + f4);
    *(v4h*)(Qs + ((r >> 4) * 4 + (f4 >> 5)) * 512 + a_sw(r & 15, f4 & 31)) = cvt4(x);
  }
  if (tid < 32) { mrow[tid] = -1e30f; lrow[tid] = 0.f; }

  // O accumulator in registers: 2m x 8n tiles; this wave owns (mtO, ntO0..+1)
  int mtO  = wave >> 2;
  int ntO0 = (wave & 3) * 2;
  v8f o0 = v8f_zero(), o1 = v8f_zero();

  int row8 = tid >> 3;   // softmax: 8 threads per query row
  int t8   = tid & 7;
  int j0   = t8 * 8;

  for (int kb = 0; kb <= qb; ++kb) {
    __syncthreads();   // protect KsT/Vs vs previous P@V
    for (int i = tid; i < 2048; i += 256) {
      int r = i >> 5, f4 = (i & 31) * 4;
      const float* rb = base + (size_t)(kb * 64 + r) * (3 * HH);
      v4f kx = *(const v4f*)(rb + HH + f4);
      v4f vx = *(const v4f*)(rb + 2 * HH + f4);
      // K^T as B fragments: group (nt = r>>4, kc = f>>5); contiguous slots
      *(v4h*)(KsT + ((r >> 4) * 4 + (f4 >> 5)) * 512 + b_sw(f4 & 31, r & 15)) = cvt4(kx);
      // V as B fragments: group (nt = f>>4, kc2 = r>>5); lane-strided stores
#pragma unroll
      for (int j = 0; j < 4; ++j) {
        int f = f4 + j;
        Vs[((f >> 4) * 2 + (r >> 5)) * 512 + b_sw(r & 31, f & 15)] = (_Float16)vx[j];
      }
    }
    __syncthreads();

    // S = Q K^T * scale : 2m x 4n tiles of 16x16, one per wave
    {
      int mt = wave >> 2;
      int nt = wave & 3;
      v8f s = v8f_zero();
#pragma unroll
      for (int kc = 0; kc < 4; ++kc) {
        v16h a = load_frag(Qs + (mt * 4 + kc) * 512);
        v16h b = load_frag(KsT + (nt * 4 + kc) * 512);
        s = wmma16(a, b, s);
      }
#pragma unroll
      for (int r = 0; r < 8; ++r)
        Ss[(mt * 16 + r + 8 * lhi) * 64 + nt * 16 + ln] = s[r] * scale;
    }
    __syncthreads();

    // online softmax update (8 threads/row, 8 cols each); cache S in regs
    bool diag = (kb == qb);
    int grow = qb * 32 + row8;
    float sv[8];
    {
      v4f s0 = *(const v4f*)(Ss + row8 * 64 + j0);
      v4f s1 = *(const v4f*)(Ss + row8 * 64 + j0 + 4);
#pragma unroll
      for (int j = 0; j < 4; ++j) { sv[j] = s0[j]; sv[4 + j] = s1[j]; }
    }
    float lm = -1e30f;
#pragma unroll
    for (int j = 0; j < 8; ++j) {
      if (diag && (kb * 64 + j0 + j) > grow) sv[j] = -1e30f;
      lm = fmaxf(lm, sv[j]);
    }
    part[row8 * 8 + t8] = lm;
    __syncthreads();
    if (t8 == 0) {
      float m = mrow[row8];
#pragma unroll
      for (int j = 0; j < 8; ++j) m = fmaxf(m, part[row8 * 8 + j]);
      float sc = __expf(mrow[row8] - m);
      srow[row8] = sc;
      lrow[row8] *= sc;
      mrow[row8] = m;
    }
    __syncthreads();
    float mnew = mrow[row8];
    float lsum = 0.f;
    v8h p8;
#pragma unroll
    for (int j = 0; j < 8; ++j) {
      float p = __expf(sv[j] - mnew);   // masked entries underflow to 0
      p8[j] = (_Float16)p;
      lsum += p;
    }
    // P as A fragments: 8-aligned j-run is one contiguous 16B slot run
    *(v8h*)(Ps + ((row8 >> 4) * 2 + (j0 >> 5)) * 512 + a_sw(row8 & 15, j0 & 31)) = p8;
    part[row8 * 8 + t8] = lsum;
    __syncthreads();
    if (t8 == 0) {
      float l = 0.f;
#pragma unroll
      for (int j = 0; j < 8; ++j) l += part[row8 * 8 + j];
      lrow[row8] += l;
    }

    // rescale O registers by srow, then O += P @ V
#pragma unroll
    for (int r = 0; r < 8; ++r) {
      float sc = srow[mtO * 16 + r + 8 * lhi];
      o0[r] *= sc;
      o1[r] *= sc;
    }
#pragma unroll
    for (int kc = 0; kc < 2; ++kc) {
      v16h a  = load_frag(Ps + (mtO * 2 + kc) * 512);
      v16h b0 = load_frag(Vs + (ntO0 * 2 + kc) * 512);
      v16h b1 = load_frag(Vs + ((ntO0 + 1) * 2 + kc) * 512);
      o0 = wmma16(a, b0, o0);
      o1 = wmma16(a, b1, o1);
    }
  }
  __syncthreads();

  // out = O / l
#pragma unroll
  for (int r = 0; r < 8; ++r) {
    int rl = mtO * 16 + r + 8 * lhi;
    size_t gr = (size_t)(qb * 32 + rl);
    float inv = 1.0f / lrow[rl];
    out[(size_t)t * NN * HH + gr * HH + ntO0 * 16 + ln]       = o0[r] * inv;
    out[(size_t)t * NN * HH + gr * HH + (ntO0 + 1) * 16 + ln] = o1[r] * inv;
  }
}

// ---------------------------------------------------------------------------
// Fused FFN: C = relu(A @ W1^T + b1) @ W2^T + b2, chunked over DFF (32-wide)
// A: [M,128] f32, W1T: [128,DFF] f16, W2T: [DFF,128] f16, C: [M,128] f32
// ---------------------------------------------------------------------------
__global__ void k_ff(const float* __restrict__ Ain, const _Float16* __restrict__ W1T,
                     const float* __restrict__ b1, const _Float16* __restrict__ W2T,
                     const float* __restrict__ b2, float* __restrict__ Cout) {
  __shared__ alignas(32) _Float16 As[16 * 512];   // A: (mt 0..3) x (kc 0..3)
  __shared__ alignas(32) _Float16 W1s[8 * 512];   // B: (ntU 0..1) x (kc 0..3)
  __shared__ alignas(32) _Float16 W2s[8 * 512];   // B: (nt 0..7), K=32
  __shared__ alignas(32) _Float16 Us[4 * 512];    // A: (mt 0..3), K=32

  int tid = threadIdx.x;
  size_t row0 = (size_t)blockIdx.x * 64;
  for (int i = tid; i < 2048; i += 256) {
    int r = i >> 5, f4 = (i & 31) * 4;
    v4f x = *(const v4f*)(Ain + (row0 + r) * 128 + f4);
    *(v4h*)(As + ((r >> 4) * 4 + (f4 >> 5)) * 512 + a_sw(r & 15, f4 & 31)) = cvt4(x);
  }
  int wave = tid >> 5, lane = tid & 31;
  int ln = lane & 15, lhi = lane >> 4;
  int mt  = wave >> 1;   // row tile (shared by U and acc phases)
  int ntU = wave & 1;    // U col tile

  v8f acc[4];
#pragma unroll
  for (int j = 0; j < 4; ++j) acc[j] = v8f_zero();

  for (int ck = 0; ck < DFF_; ck += 32) {
    __syncthreads();
    for (int i = tid; i < 512; i += 256) {
      // W1s: [128 k][32 c], v8h over c
      int k = i >> 2, c8 = (i & 3) * 8;
      v8h x = *(const v8h*)(W1T + (size_t)k * DFF_ + ck + c8);
#pragma unroll
      for (int j = 0; j < 8; ++j) {
        int c = c8 + j;
        W1s[((c >> 4) * 4 + (k >> 5)) * 512 + b_sw(k & 31, c & 15)] = x[j];
      }
      // W2s: [32 k2][128 c], v8h over c
      int k2 = i >> 4, c8b = (i & 15) * 8;
      v8h y = *(const v8h*)(W2T + (size_t)(ck + k2) * 128 + c8b);
#pragma unroll
      for (int j = 0; j < 8; ++j) {
        int c = c8b + j;
        W2s[(c >> 4) * 512 + b_sw(k2, c & 15)] = y[j];
      }
    }
    __syncthreads();

    // U = relu(A @ W1s + b1[ck..ck+32))  — 4m x 2n tiles, one per wave
    v8f u = v8f_zero();
#pragma unroll
    for (int kc = 0; kc < 4; ++kc) {
      v16h a = load_frag(As + (mt * 4 + kc) * 512);
      v16h b = load_frag(W1s + (ntU * 4 + kc) * 512);
      u = wmma16(a, b, u);
    }
    float bb = b1[ck + ntU * 16 + ln];
#pragma unroll
    for (int r = 0; r < 8; ++r)
      Us[mt * 512 + a_sw(r + 8 * lhi, ntU * 16 + ln)] =
          (_Float16)fmaxf(u[r] + bb, 0.f);
    __syncthreads();

    // acc += U @ W2s — each wave 4 tiles in its row-tile, nt = (wave&1)*4 + j
    v16h a2 = load_frag(Us + mt * 512);
#pragma unroll
    for (int j = 0; j < 4; ++j) {
      int nt = (wave & 1) * 4 + j;
      v16h b = load_frag(W2s + nt * 512);
      acc[j] = wmma16(a2, b, acc[j]);
    }
  }
  __syncthreads();

#pragma unroll
  for (int j = 0; j < 4; ++j) {
    int col = ((wave & 1) * 4 + j) * 16 + ln;
    float bb = b2[col];
#pragma unroll
    for (int r = 0; r < 8; ++r) {
      size_t row = row0 + mt * 16 + r + 8 * lhi;
      Cout[row * 128 + col] = acc[j][r] + bb;
    }
  }
}

// ---------------------------------------------------------------------------
// LayerNorm over last dim (128), optional fused residual: out = LN(x (+res))
// One wave (32 lanes) per row, 8 rows per 256-thread block, float4 I/O.
// ---------------------------------------------------------------------------
template <bool RES>
__global__ void k_ln(const float* __restrict__ x, const float* __restrict__ res,
                     const float* __restrict__ g, const float* __restrict__ b,
                     float* __restrict__ out, int rows) {
  int row = blockIdx.x * 8 + (threadIdx.x >> 5);
  if (row >= rows) return;
  int lane = threadIdx.x & 31;
  int c = lane * 4;
  v4f v = *(const v4f*)(x + (size_t)row * 128 + c);
  if (RES) v += *(const v4f*)(res + (size_t)row * 128 + c);
  float s = v[0] + v[1] + v[2] + v[3];
#pragma unroll
  for (int off = 16; off > 0; off >>= 1) s += __shfl_xor(s, off, 32);
  float mu = s * (1.f / 128.f);
  float var = 0.f;
#pragma unroll
  for (int j = 0; j < 4; ++j) { float d = v[j] - mu; var += d * d; }
#pragma unroll
  for (int off = 16; off > 0; off >>= 1) var += __shfl_xor(var, off, 32);
  float inv = rsqrtf(var * (1.f / 128.f) + 1e-5f);
  v4f gg = *(const v4f*)(g + c);
  v4f bb = *(const v4f*)(b + c);
  v4f o;
#pragma unroll
  for (int j = 0; j < 4; ++j) o[j] = (v[j] - mu) * inv * gg[j] + bb[j];
  *(v4f*)(out + (size_t)row * 128 + c) = o;
}

// ---------------------------------------------------------------------------
// Host launch
// ---------------------------------------------------------------------------
extern "C" void kernel_launch(void* const* d_in, const int* in_sizes, int n_in,
                              void* d_out, int out_size, void* d_ws, size_t ws_size,
                              hipStream_t stream) {
  (void)in_sizes; (void)n_in; (void)out_size; (void)ws_size;

  const float* x_in   = (const float*)d_in[0];
  const int*   ei     = (const int*)d_in[1];
  const float* conv_W = (const float*)d_in[2];
  const float* conv_b = (const float*)d_in[3];
  const float* Wqkv   = (const float*)d_in[4];
  const float* bqkv   = (const float*)d_in[5];
  const float* Wo     = (const float*)d_in[6];
  const float* bo     = (const float*)d_in[7];
  const float* ln1_g  = (const float*)d_in[8];
  const float* ln1_b  = (const float*)d_in[9];
  const float* W1     = (const float*)d_in[10];
  const float* b1     = (const float*)d_in[11];
  const float* W2     = (const float*)d_in[12];
  const float* b2     = (const float*)d_in[13];
  const float* ln2_g  = (const float*)d_in[14];
  const float* ln2_b  = (const float*)d_in[15];

  const int* src = ei;
  const int* dst = ei + EE;

  // workspace carve-up
  char* ws = (char*)d_ws;
  size_t off = 0;
  auto alloc = [&](size_t bytes) -> void* {
    void* p = ws + off;
    off = (off + bytes + 255) & ~(size_t)255;
    return p;
  };
  const size_t fN = (size_t)NN * TT * HH;  // 4,194,304 elems
  float* bufA   = (float*)alloc(fN * 4);
  float* bufB   = (float*)alloc(fN * 4);
  float* bufY   = (float*)alloc(fN * 4);
  float* bufX   = (float*)alloc(fN * 4);
  float* bufQKV = (float*)alloc((size_t)TT * NN * 3 * HH * 4);
  float* dinv   = (float*)alloc((size_t)NN * 4);
  _Float16* cW16    = (_Float16*)alloc((size_t)HH * HH * 2);
  _Float16* WqkvT16 = (_Float16*)alloc((size_t)HH * 3 * HH * 2);
  _Float16* WoT16   = (_Float16*)alloc((size_t)HH * HH * 2);
  _Float16* W1T16   = (_Float16*)alloc((size_t)HH * DFF_ * 2);
  _Float16* W2T16   = (_Float16*)alloc((size_t)DFF_ * HH * 2);

  const int thr = 256;

  // GCN degree normalization (edge_index is static across layers)
  k_deg_init <<<(NN + thr - 1) / thr, thr, 0, stream>>>(dinv, NN);
  k_deg_edges<<<(EE + thr - 1) / thr, thr, 0, stream>>>(dst, dinv, EE);
  k_deg_rsqrt<<<(NN + thr - 1) / thr, thr, 0, stream>>>(dinv, NN);

  for (int l = 0; l < LL; ++l) {
    const float* xin = (l == 0) ? x_in : bufX;

    // convert weights of this layer to f16 in GEMM-friendly [K,N] layouts
    k_convert_f16  <<<(HH * HH + thr - 1) / thr, thr, 0, stream>>>(conv_W + (size_t)l * HH * HH, cW16, HH * HH);
    k_transpose_f16<<<(3 * HH * HH + thr - 1) / thr, thr, 0, stream>>>(Wqkv + (size_t)l * 3 * HH * HH, WqkvT16, 3 * HH, HH);
    k_transpose_f16<<<(HH * HH + thr - 1) / thr, thr, 0, stream>>>(Wo + (size_t)l * HH * HH, WoT16, HH, HH);
    k_transpose_f16<<<(DFF_ * HH + thr - 1) / thr, thr, 0, stream>>>(W1 + (size_t)l * DFF_ * HH, W1T16, DFF_, HH);
    k_transpose_f16<<<(HH * DFF_ + thr - 1) / thr, thr, 0, stream>>>(W2 + (size_t)l * HH * DFF_, W2T16, HH, DFF_);

    // --- GCN linear: h_lin = x @ conv_W  ([N*T,128] x [128,128]) ---
    {
      dim3 g(HH / 64, (NN * TT) / 64, 1);
      k_gemm<false, false><<<g, thr, 0, stream>>>(xin, 0, HH, cW16, nullptr,
                                                  nullptr, 0, bufA, 0, HH,
                                                  NN * TT, HH, HH);
    }
    // --- aggregation ---
    k_agg_self <<<(int)(fN / 4 / thr), thr, 0, stream>>>(bufA, dinv, bufB);
    k_agg_edges<<<EE, thr, 0, stream>>>(bufA, src, dst, dinv, bufB);
    // --- bias + relu + *sqrt(H) + PE, transpose to y [T,N,H] ---
    k_relu_pe_y<<<(int)(fN / 4 / thr), thr, 0, stream>>>(bufB, conv_b + (size_t)l * HH, bufY);

    // --- QKV projection, batched over T ---
    {
      dim3 g((3 * HH) / 64, NN / 64, TT);
      k_gemm<true, false><<<g, thr, 0, stream>>>(bufY, (long long)NN * HH, HH,
                                                 WqkvT16, bqkv + (size_t)l * 3 * HH,
                                                 nullptr, 0, bufQKV,
                                                 (long long)NN * 3 * HH, 3 * HH,
                                                 NN, 3 * HH, HH);
    }
    // --- causal flash attention over nodes ---
    {
      dim3 g(NN / 32, TT, 1);
      k_attn<<<g, thr, 0, stream>>>(bufQKV, bufA);
    }
    // --- output projection + residual add (res1 = y + attn@Wo^T + bo) ---
    {
      dim3 g(HH / 64, NN / 64, TT);
      k_gemm<true, true><<<g, thr, 0, stream>>>(bufA, (long long)NN * HH, HH,
                                                WoT16, bo + (size_t)l * HH, bufY,
                                                (long long)NN * HH, bufB,
                                                (long long)NN * HH, HH, NN, HH, HH);
    }
    // --- LN1: y1 = LN(res1) ---
    k_ln<false><<<(NN * TT) / 8, thr, 0, stream>>>(bufB, nullptr,
                                                   ln1_g + (size_t)l * HH,
                                                   ln1_b + (size_t)l * HH, bufA,
                                                   NN * TT);
    // --- fused FFN: ff = relu(y1@W1^T+b1)@W2^T + b2 ---
    k_ff<<<(NN * TT) / 64, thr, 0, stream>>>(bufA, W1T16, b1 + (size_t)l * DFF_,
                                             W2T16, b2 + (size_t)l * HH, bufB);
    // --- LN2: y2 = LN(y1 + ff) ---
    k_ln<true><<<(NN * TT) / 8, thr, 0, stream>>>(bufB, bufA,
                                                  ln2_g + (size_t)l * HH,
                                                  ln2_b + (size_t)l * HH, bufY,
                                                  NN * TT);
    // --- transpose back to [N,T,H] for next layer / output ---
    float* xdst = (l == LL - 1) ? (float*)d_out : bufX;
    k_trans_to_x<<<(int)(fN / 4 / thr), thr, 0, stream>>>(bufY, xdst);
  }
}
